// RNN_70446053589049
// MI455X (gfx1250) — matrix operand order repaired and checked by reading, not compiled, not run on hardware
//
#include <hip/hip_runtime.h>
#include <hip/hip_bf16.h>

// Problem sizes (match reference)
#define BB    128
#define TT    2048
#define DDIM  128
#define HH    512
#define LCLS  10

#define GH    (4*HH)        // 2048 gate rows
#define KK    (DDIM+HH)     // 640 combined K (x then h)
#define NSL   16            // hidden slices
#define SH    (HH/NSL)      // 32 hidden units per slice
#define ROWS  (4*SH)        // 128 gate rows per workgroup
#define NBT   8             // batch tiles
#define MB    (BB/NBT)      // 16 batch rows per tile
#define KT    (KK/32)       // 20 WMMA k-steps
#define WSTR  (KK+8)        // 648 bf16: padded LDS row stride (bank spread)
#define GSTR  20            // padded f32 gate-buffer stride

typedef __attribute__((ext_vector_type(16))) __bf16       v16bf;
typedef __attribute__((ext_vector_type(8)))  float        v8f;
typedef __attribute__((ext_vector_type(4)))  unsigned int u32x4;

union ABFrag { v16bf v; u32x4 q[2]; };
union CAcc   { v8f v; float f[8]; };

// Workspace layout (bytes)
#define SZ_XBF  ((size_t)BB*TT*DDIM*2)        // x in bf16: 64 MiB
#define OFF_XBF ((size_t)0)
#define SZ_WC   ((size_t)GH*KK*2)             // combined weights bf16: 2.5 MiB
#define OFF_WC  (OFF_XBF + SZ_XBF)
#define SZ_H    ((size_t)2*BB*HH*2)           // double-buffered h (bf16)
#define OFF_H   (OFF_WC + SZ_WC)
#define SZ_LG   ((size_t)BB*LCLS*4)           // f32 logits accumulator
#define OFF_LG  (OFF_H + SZ_H)
#define OFF_CNT (OFF_LG + SZ_LG)              // barrier counters (128 u32)

#define SMEM_BYTES ((size_t)ROWS*WSTR*2 + (size_t)ROWS*GSTR*4 + (size_t)ROWS*4 + (size_t)MB*33*4)

static __device__ __forceinline__ unsigned short f2bf(float f) {
  unsigned int x = __float_as_uint(f);
  x += 0x7fffu + ((x >> 16) & 1u);          // round-to-nearest-even
  return (unsigned short)(x >> 16);
}
static __device__ __forceinline__ float sigf(float x) {
  return 1.f / (1.f + __expf(-x));
}
static __device__ __forceinline__ float tanh_fast(float x) {
  float e = __expf(-2.f * fabsf(x));
  float r = (1.f - e) / (1.f + e);
  return x < 0.f ? -r : r;
}

// ---------------------------------------------------------------------------
// Kernel 0: convert x -> bf16, pack Wc = [W_ih | W_hh] bf16, zero h/logits/cnt
// ---------------------------------------------------------------------------
__global__ void lstm_prep(const float* __restrict__ x,
                          const float* __restrict__ W_ih,
                          const float* __restrict__ W_hh,
                          unsigned short* __restrict__ xbf,
                          unsigned short* __restrict__ wcw,
                          unsigned short* __restrict__ hbuf,
                          float* __restrict__ logits,
                          unsigned int* __restrict__ cnt) {
  const size_t n_x = (size_t)BB * TT * DDIM;
  const size_t n_w = (size_t)GH * KK;
  const size_t n_h = (size_t)2 * BB * HH;
  const size_t n_l = (size_t)BB * LCLS;
  const size_t n_c = 128;
  const size_t total = n_x + n_w + n_h + n_l + n_c;
  const size_t stride = (size_t)gridDim.x * blockDim.x;
  for (size_t i = (size_t)blockIdx.x * blockDim.x + threadIdx.x; i < total; i += stride) {
    if (i < n_x) {
      xbf[i] = f2bf(x[i]);
    } else if (i < n_x + n_w) {
      size_t j = i - n_x;
      size_t g = j / KK, c = j % KK;
      float v = (c < DDIM) ? W_ih[g * DDIM + c] : W_hh[g * HH + (c - DDIM)];
      wcw[j] = f2bf(v);
    } else if (i < n_x + n_w + n_h) {
      hbuf[i - n_x - n_w] = 0;
    } else if (i < n_x + n_w + n_h + n_l) {
      logits[i - n_x - n_w - n_h] = 0.f;
    } else {
      cnt[i - n_x - n_w - n_h - n_l] = 0u;
    }
  }
}

// ---------------------------------------------------------------------------
// Kernel 1: persistent LSTM recurrence.
// Grid = 8 batch tiles x 16 hidden slices. Weights LDS-resident (160 KB/WGP).
// Per step: gates(16x128) = [x_t | h] (16x640) @ Wc_slice^T via 20 bf16 WMMAs
// per wave; gate combine in LDS; h exchanged via global double buffer with a
// per-tile-group atomic barrier.
// ---------------------------------------------------------------------------
__global__ void lstm_recurrent(const unsigned short* __restrict__ xbf,
                               const unsigned short* __restrict__ wcw,
                               const float* __restrict__ bias,
                               const float* __restrict__ W_out,
                               unsigned short* __restrict__ hbuf,
                               float* __restrict__ logits,
                               unsigned int* __restrict__ cnt) {
  extern __shared__ char smem[];
  unsigned short* wlds = (unsigned short*)smem;                 // ROWS x WSTR bf16
  float* gbuf = (float*)(smem + (size_t)ROWS * WSTR * 2);       // ROWS x GSTR f32
  float* blds = gbuf + ROWS * GSTR;                             // ROWS f32 bias
  float* hfin = blds + ROWS;                                    // MB x 33 f32

  const int sl = blockIdx.x % NSL;   // hidden slice
  const int bt = blockIdx.x / NSL;   // batch tile
  const int m0 = bt * MB;
  const int tid  = threadIdx.x;
  const int wave = tid >> 5;
  const int lane = tid & 31;

  // Stage this slice's 128 gate rows of Wc into LDS (once; reused 2048 steps).
  for (int idx = tid; idx < ROWS * (KK / 8); idx += 256) {
    int r  = idx / (KK / 8);
    int ch = idx % (KK / 8);
    int q = r / SH, u = r % SH;
    int grow = q * HH + sl * SH + u;          // global gate row (i,f,g,o blocks)
    *(u32x4*)(wlds + (size_t)r * WSTR + ch * 8) =
        *(const u32x4*)(wcw + (size_t)grow * KK + ch * 8);
  }
  for (int r = tid; r < ROWS; r += 256) {
    int q = r / SH, u = r % SH;
    blds[r] = bias[q * HH + sl * SH + u];
  }
  __syncthreads();

  // WMMA fragment lane geometry (16-bit A 16x32 / B 32x16 layouts, ISA 7.12.2)
  const int lrow  = lane & 15;
  const int mhalf = (lane >> 4) * 8;    // C rows M offset
  const int kaoff = (lane >> 4) * 8;    // A: K chunks {0..7,16..23} vs {8..15,24..31}
  const int kboff = (lane >> 4) * 8;    // B: K chunks {0..7,16..23} vs {8..15,24..31}
  const int nloc  = wave * 16 + lrow;   // gate row owned by this lane's column
  const unsigned short* wrow = wlds + (size_t)nloc * WSTR;
  const unsigned short* xrow = xbf + (size_t)(m0 + lrow) * TT * DDIM;

  float cc0 = 0.f, cc1 = 0.f;           // cell state: 2 cells per thread
  const int e0 = tid * 2;
  const int em = e0 >> 5;               // batch row within tile
  const int eu = e0 & 31;               // hidden unit within slice (even)

  unsigned int* bar = &cnt[bt * 16];

  for (int t = 0; t < TT; ++t) {
    const int p = t & 1;
    const unsigned short* hsrc = hbuf + (size_t)p * BB * HH;
    unsigned short* hdst = hbuf + (size_t)(p ^ 1) * BB * HH;

    v8f acc = {0.f, 0.f, 0.f, 0.f, 0.f, 0.f, 0.f, 0.f};
    const unsigned short* xa = xrow + (size_t)t * DDIM + kaoff;
    const unsigned short* ha = hsrc + (size_t)(m0 + lrow) * HH + kaoff;
    #pragma unroll
    for (int ks = 0; ks < KT; ++ks) {
      ABFrag a, b;
      const unsigned short* asrc =
          (ks < DDIM / 32) ? (xa + ks * 32) : (ha + (ks - DDIM / 32) * 32);
      a.q[0] = *(const u32x4*)(asrc);
      a.q[1] = *(const u32x4*)(asrc + 16);
      const unsigned short* bsrc = wrow + ks * 32 + kboff;
      b.q[0] = *(const u32x4*)(bsrc);
      b.q[1] = *(const u32x4*)(bsrc + 16);
      acc = __builtin_amdgcn_wmma_f32_16x16x32_bf16(false, a.v, false, b.v,
                                                    (short)0, acc, false, false);
    }
    if (t + 1 < TT)  // pull next timestep's x tile toward the WGP
      __builtin_prefetch((const void*)(xrow + (size_t)(t + 1) * DDIM + kaoff), 0, 0);

    // Spill C (gates, f32) to LDS as gbuf[gate_row][batch]
    {
      CAcc cfr; cfr.v = acc;
      float* gdst = gbuf + nloc * GSTR + mhalf;
      #pragma unroll
      for (int v = 0; v < 8; ++v) gdst[v] = cfr.f[v];
    }
    __syncthreads();

    // Gate combine (i,f,g,o live in different waves' tiles): 2 cells/thread
    float hv0, hv1;
    {
      int u = eu;
      float gi = gbuf[(0 * SH + u) * GSTR + em] + blds[0 * SH + u];
      float gf = gbuf[(1 * SH + u) * GSTR + em] + blds[1 * SH + u];
      float gg = gbuf[(2 * SH + u) * GSTR + em] + blds[2 * SH + u];
      float go = gbuf[(3 * SH + u) * GSTR + em] + blds[3 * SH + u];
      cc0 = sigf(gf) * cc0 + sigf(gi) * tanh_fast(gg);
      hv0 = sigf(go) * tanh_fast(cc0);
    }
    {
      int u = eu + 1;
      float gi = gbuf[(0 * SH + u) * GSTR + em] + blds[0 * SH + u];
      float gf = gbuf[(1 * SH + u) * GSTR + em] + blds[1 * SH + u];
      float gg = gbuf[(2 * SH + u) * GSTR + em] + blds[2 * SH + u];
      float go = gbuf[(3 * SH + u) * GSTR + em] + blds[3 * SH + u];
      cc1 = sigf(gf) * cc1 + sigf(gi) * tanh_fast(gg);
      hv1 = sigf(go) * tanh_fast(cc1);
    }
    union { unsigned short su[2]; unsigned int w; } pk;
    pk.su[0] = f2bf(hv0);
    pk.su[1] = f2bf(hv1);
    *(unsigned int*)(hdst + (size_t)(m0 + em) * HH + sl * SH + eu) = pk.w;
    if (t == TT - 1) {
      hfin[em * 33 + eu]     = hv0;
      hfin[em * 33 + eu + 1] = hv1;
    }

    // Release h, then barrier across the 16 WGs of this batch-tile group.
    __threadfence();
    __syncthreads();
    if (tid == 0) {
      atomicAdd(bar, 1u);
      const unsigned int tgt = (unsigned int)(t + 1) * NSL;
      while (__hip_atomic_load(bar, __ATOMIC_RELAXED, __HIP_MEMORY_SCOPE_AGENT) < tgt)
        __builtin_amdgcn_s_sleep(2);
    }
    __syncthreads();
  }

  // Partial logits from this slice's h_final (16 x 32) into global f32 accumulator
  for (int idx = tid; idx < MB * LCLS; idx += 256) {
    int m = idx / LCLS, l = idx % LCLS;
    float sum = 0.f;
    #pragma unroll
    for (int u = 0; u < SH; ++u)
      sum += hfin[m * 33 + u] * W_out[(size_t)l * HH + sl * SH + u];
    atomicAdd(&logits[(size_t)(m0 + m) * LCLS + l], sum);
  }
}

// ---------------------------------------------------------------------------
// Kernel 2: softmax over (128 x 10) logits
// ---------------------------------------------------------------------------
__global__ void softmax10(const float* __restrict__ logits, float* __restrict__ out) {
  int b = threadIdx.x;
  if (b < BB) {
    float v[LCLS];
    float mx = -1e30f;
    #pragma unroll
    for (int l = 0; l < LCLS; ++l) { v[l] = logits[b * LCLS + l]; mx = fmaxf(mx, v[l]); }
    float sum = 0.f;
    #pragma unroll
    for (int l = 0; l < LCLS; ++l) { v[l] = __expf(v[l] - mx); sum += v[l]; }
    float inv = 1.f / sum;
    #pragma unroll
    for (int l = 0; l < LCLS; ++l) out[b * LCLS + l] = v[l] * inv;
  }
}

extern "C" void kernel_launch(void* const* d_in, const int* in_sizes, int n_in,
                              void* d_out, int out_size, void* d_ws, size_t ws_size,
                              hipStream_t stream) {
  (void)in_sizes; (void)n_in; (void)out_size; (void)ws_size;
  const float* x     = (const float*)d_in[0];
  const float* W_ih  = (const float*)d_in[1];
  const float* W_hh  = (const float*)d_in[2];
  const float* bias  = (const float*)d_in[3];
  const float* W_out = (const float*)d_in[4];

  char* ws = (char*)d_ws;
  unsigned short* xbf = (unsigned short*)(ws + OFF_XBF);
  unsigned short* wcw = (unsigned short*)(ws + OFF_WC);
  unsigned short* hb  = (unsigned short*)(ws + OFF_H);
  float* logits       = (float*)(ws + OFF_LG);
  unsigned int* cnt   = (unsigned int*)(ws + OFF_CNT);

  lstm_prep<<<2048, 256, 0, stream>>>(x, W_ih, W_hh, xbf, wcw, hb, logits, cnt);

  static_assert(SMEM_BYTES <= 320u * 1024u, "LDS budget exceeded");
  (void)hipFuncSetAttribute((const void*)lstm_recurrent,
                            hipFuncAttributeMaxDynamicSharedMemorySize,
                            (int)SMEM_BYTES);
  lstm_recurrent<<<dim3(NBT * NSL), dim3(256), SMEM_BYTES, stream>>>(
      xbf, wcw, bias, W_out, hb, logits, cnt);

  softmax10<<<1, 128, 0, stream>>>(logits, (float*)d_out);
}
